// WaterLevelLSTM_17617955848226
// MI455X (gfx1250) — compile-verified, matrix operand order checked by reading, not run on hardware
//
#include <hip/hip_runtime.h>

typedef _Float16 v16h __attribute__((ext_vector_type(16)));
typedef _Float16 v8h  __attribute__((ext_vector_type(8)));
typedef float    v8f  __attribute__((ext_vector_type(8)));
typedef float    v4f  __attribute__((ext_vector_type(4)));

constexpr int kH = 64;   // hidden size
constexpr int kT = 5;    // timesteps
constexpr int kWpack = 16 * 2 * 32 * 16;   // packed W_hh halves (16384)

// tanh on the hardware TRANS pipe if gfx1250 exposes V_TANH_F32, else exp2/rcp
#if __has_builtin(__builtin_amdgcn_tanhf)
__device__ __forceinline__ float hw_tanh(float x) { return __builtin_amdgcn_tanhf(x); }
#else
__device__ __forceinline__ float hw_tanh(float x) {
    float e = __builtin_amdgcn_exp2f(2.8853900817779268f * x);
    return 1.0f - 2.0f * __builtin_amdgcn_rcpf(1.0f + e);
}
#endif
__device__ __forceinline__ float hw_sigmoid(float x) {
    return __builtin_fmaf(0.5f, hw_tanh(0.5f * x), 0.5f);
}

__device__ __forceinline__ v16h cat8(v8h lo, v8h hi) {
    return __builtin_shufflevector(lo, hi, 0,1,2,3,4,5,6,7,8,9,10,11,12,13,14,15);
}

// ---------------------------------------------------------------------------
// Pre-pack W_hh (f32, [256 x 64]) into f16 B-fragment byte order in d_ws.
// Fragment (nt, kk), lane L: N = L&15, Kb = 32*kk + 8*(L>>4);
//   elem e<8  -> K = Kb + e
//   elem e>=8 -> K = Kb + 16 + (e-8)
// value = W_hh[(16*nt + N)*64 + K]
// ---------------------------------------------------------------------------
__global__ __launch_bounds__(256)
void pack_whh_kernel(const float* __restrict__ W_hh, _Float16* __restrict__ wsW) {
    const int idx  = blockIdx.x * 256 + threadIdx.x;   // 64 blocks -> 16384
    const int e    = idx & 15;
    const int lane = (idx >> 4) & 31;
    const int kk   = (idx >> 9) & 1;
    const int nt   = idx >> 10;
    const int l16  = lane & 15;
    const int half = lane >> 4;
    const int Kb   = kk * 32 + half * 8;
    const int K    = Kb + (e < 8 ? e : e + 8);
    wsW[idx] = (_Float16)W_hh[(nt * 16 + l16) * kH + K];
}

// ---------------------------------------------------------------------------
// Main LSTM kernel. PRE: weights come pre-packed from d_ws; else build inline.
// ---------------------------------------------------------------------------
template <bool PRE>
__global__ __launch_bounds__(128)
void lstm_wmma_v3(const float* __restrict__ x,
                  const float* __restrict__ W_ih,
                  const float* __restrict__ W_hh,
                  const float* __restrict__ b_ih,
                  const float* __restrict__ b_hh,
                  const float* __restrict__ W_fc,
                  const float* __restrict__ b_fc,
                  const _Float16* __restrict__ wsW,
                  float* __restrict__ out)
{
    // per-wave h staging tile: 4 waves x (16 rows x 64 k) f16 = 8 KB
    __shared__ __align__(16) _Float16 sH[4 * 16 * kH];

    const int tid  = threadIdx.x;
    const int lane = tid & 31;
    const int wave = tid >> 5;
    const int half = lane >> 4;     // 0 = lanes 0-15, 1 = lanes 16-31
    const int l16  = lane & 15;

    _Float16* hbuf = sH + wave * (16 * kH);
    const int m0 = (blockIdx.x * 4 + wave) * 16;   // batch tile base

    // ---- W_hh B-fragments, register resident (16 N-tiles x 2 K-steps) ----
    v16h bfr[16][2];
    #pragma unroll
    for (int nt = 0; nt < 16; ++nt) {
        #pragma unroll
        for (int kk = 0; kk < 2; ++kk) {
            if constexpr (PRE) {
                // packed layout: coalesced, aligned, no conversion
                bfr[nt][kk] = *(const v16h*)(wsW + ((nt * 2 + kk) * 32 + lane) * 16);
            } else {
                const int Kb = kk * 32 + half * 8;
                const float* wrow = W_hh + (nt * 16 + l16) * kH + Kb;
                v4f f0 = *(const v4f*)(wrow);
                v4f f1 = *(const v4f*)(wrow + 4);
                v4f f2 = *(const v4f*)(wrow + 16);
                v4f f3 = *(const v4f*)(wrow + 20);
                v16h b;
                #pragma unroll
                for (int j = 0; j < 4; ++j) {
                    b[j]      = (_Float16)f0[j];
                    b[4 + j]  = (_Float16)f1[j];
                    b[8 + j]  = (_Float16)f2[j];
                    b[12 + j] = (_Float16)f3[j];
                }
                bfr[nt][kk] = b;
            }
        }
    }

    // per-lane gate-column constants: column = 16*nt + l16
    float wih_l[16], bias_l[16];
    #pragma unroll
    for (int nt = 0; nt < 16; ++nt) {
        const int gc = nt * 16 + l16;
        wih_l[nt]  = W_ih[gc];
        bias_l[nt] = b_ih[gc] + b_hh[gc];
    }

    float c[4][8];   // cell state, D-tile layout: lane = N(hidden), reg = M(batch)

    // ---- t = 0 peeled: h0 == 0, c0 == 0 -> gates = xg only, no WMMA ----
    {
        float xv[8];
        #pragma unroll
        for (int r = 0; r < 8; ++r)
            xv[r] = x[(size_t)(m0 + r + half * 8) * kT + 0];
        #pragma unroll
        for (int n = 0; n < 4; ++n) {
            #pragma unroll
            for (int r = 0; r < 8; ++r) {
                const float gi = __builtin_fmaf(xv[r], wih_l[n],      bias_l[n]);
                const float gg = __builtin_fmaf(xv[r], wih_l[8 + n],  bias_l[8 + n]);
                const float go = __builtin_fmaf(xv[r], wih_l[12 + n], bias_l[12 + n]);
                const float cn = hw_sigmoid(gi) * hw_tanh(gg);
                c[n][r] = cn;
                const float hn = hw_sigmoid(go) * hw_tanh(cn);
                hbuf[(r + half * 8) * kH + n * 16 + l16] = (_Float16)hn;
            }
        }
    }

    // ---- t = 1..4: WMMA recurrence (same-wave DS ops are in-order) ----
    for (int t = 1; t < kT; ++t) {
        // A fragments: h tile 16x64 f16, A layout (lane = M, regs = K)
        v16h a[2];
        #pragma unroll
        for (int kk = 0; kk < 2; ++kk) {
            const int Kb = kk * 32 + half * 8;
            const _Float16* row = hbuf + l16 * kH + Kb;
            a[kk] = cat8(*(const v8h*)row, *(const v8h*)(row + 16));
        }

        float xv[8];
        #pragma unroll
        for (int r = 0; r < 8; ++r)
            xv[r] = x[(size_t)(m0 + r + half * 8) * kT + t];

        #pragma unroll
        for (int n = 0; n < 4; ++n) {          // hidden slice: units 16n..16n+15
            v8f acc[4];
            #pragma unroll
            for (int g = 0; g < 4; ++g) {      // gates i, f, g, o
                const int nt = g * 4 + n;      // gate-column tile index
                v8f cacc;
                #pragma unroll
                for (int r = 0; r < 8; ++r)    // seed C with xg = x*W_ih + b
                    cacc[r] = __builtin_fmaf(xv[r], wih_l[nt], bias_l[nt]);
                cacc = __builtin_amdgcn_wmma_f32_16x16x32_f16(
                    false, a[0], false, bfr[nt][0], (short)0, cacc, false, false);
                cacc = __builtin_amdgcn_wmma_f32_16x16x32_f16(
                    false, a[1], false, bfr[nt][1], (short)0, cacc, false, false);
                acc[g] = cacc;
            }
            #pragma unroll
            for (int r = 0; r < 8; ++r) {
                const float ig = hw_sigmoid(acc[0][r]);
                const float fg = hw_sigmoid(acc[1][r]);
                const float gg = hw_tanh(acc[2][r]);
                const float og = hw_sigmoid(acc[3][r]);
                const float cn = __builtin_fmaf(fg, c[n][r], ig * gg);
                c[n][r] = cn;
                const float hn = og * hw_tanh(cn);
                hbuf[(r + half * 8) * kH + n * 16 + l16] = (_Float16)hn;
            }
        }
    }

    // ---- projection: out[b] = h . W_fc + b_fc (h re-read from staging) ----
    float wfc_l[4];
    #pragma unroll
    for (int n = 0; n < 4; ++n) wfc_l[n] = W_fc[n * 16 + l16];
    const float bf = b_fc[0];

    #pragma unroll
    for (int r = 0; r < 8; ++r) {
        float p = 0.0f;
        #pragma unroll
        for (int n = 0; n < 4; ++n)
            p = __builtin_fmaf((float)hbuf[(r + half * 8) * kH + n * 16 + l16],
                               wfc_l[n], p);
        // reduce over the 16 lanes of each half-wave (wave32-safe masks)
        p += __shfl_xor(p, 1);
        p += __shfl_xor(p, 2);
        p += __shfl_xor(p, 4);
        p += __shfl_xor(p, 8);
        if (l16 == 0) out[m0 + r + half * 8] = p + bf;
    }
}

extern "C" void kernel_launch(void* const* d_in, const int* in_sizes, int n_in,
                              void* d_out, int out_size, void* d_ws, size_t ws_size,
                              hipStream_t stream) {
    const float* x    = (const float*)d_in[0];
    const float* W_ih = (const float*)d_in[1];
    const float* W_hh = (const float*)d_in[2];
    const float* b_ih = (const float*)d_in[3];
    const float* b_hh = (const float*)d_in[4];
    const float* W_fc = (const float*)d_in[5];
    const float* b_fc = (const float*)d_in[6];
    float* out = (float*)d_out;

    const int B = in_sizes[0] / kT;        // x is [B, T, 1]
    const int blocks = B / 64;             // 4 waves/block * 16 batch rows/wave

    if (ws_size >= kWpack * sizeof(_Float16)) {
        _Float16* wsW = (_Float16*)d_ws;
        pack_whh_kernel<<<kWpack / 256, 256, 0, stream>>>(W_hh, wsW);
        lstm_wmma_v3<true><<<blocks, 128, 0, stream>>>(x, W_ih, W_hh, b_ih, b_hh,
                                                       W_fc, b_fc, wsW, out);
    } else {
        lstm_wmma_v3<false><<<blocks, 128, 0, stream>>>(x, W_ih, W_hh, b_ih, b_hh,
                                                        W_fc, b_fc, nullptr, out);
    }
}
